// HungarianMatcher_90555090469352
// MI455X (gfx1250) — compile-verified, matrix operand order checked by reading, not run on hardware
//
#include <hip/hip_runtime.h>

#define BS   32
#define NQ   500
#define NT   64
#define NQP  512   // padded column count (one thread per column)
#define BSZ  512   // 16 wave32 waves

#define W_BOXES  5.0f
#define W_LOGITS 2.0f
#define ANG_MIN  (-1.5707963267948966f)
#define ANG_SPAN (3.1415926535897932f)

#define PADCOST 1.0e30f
#define BIGS    3.0e38f

typedef int v8i __attribute__((ext_vector_type(8)));

#if defined(__has_builtin)
#if __has_builtin(__builtin_amdgcn_global_load_async_to_lds_b32) && \
    __has_builtin(__builtin_amdgcn_s_wait_asynccnt)
#define USE_ASYNC_LDS 1
#endif
#endif

__device__ __forceinline__ void red_pair(float& v, int& j, int off) {
  float ov = __shfl_down(v, off, 32);
  int   oj = __shfl_down(j, off, 32);
  if (ov < v || (ov == v && oj < j)) { v = ov; j = oj; }
}

constexpr size_t SMEM_WORDS =
    (size_t)NT * NQP   // cost[t][q] (transposed)
  + NQP                // v (column duals)
  + NQP                // spc (shortest path costs)
  + NT                 // u (row duals)
  + NQP                // path
  + NQP                // remaining
  + NQP                // row4col
  + NT                 // sr (rows visited)
  + NT                 // col4row
  + NT                 // rank
  + NT * 4             // staged tgt boxes
  + NT                 // staged tgt angles
  + 16 + 16            // per-wave reduction partials
  + 8;                 // broadcast slots
constexpr size_t SMEM_BYTES = SMEM_WORDS * 4;

__global__ __launch_bounds__(BSZ, 1)
void hungarian_matcher_kernel(const float* __restrict__ g_out_boxes,
                              const float* __restrict__ g_out_angle,
                              const float* __restrict__ g_out_logits,
                              const float* __restrict__ g_tgt_boxes,
                              const float* __restrict__ g_tgt_angle,
                              int* __restrict__ g_out) {
  extern __shared__ char smem_raw[];
  float* costM = (float*)smem_raw;
  float* vArr  = costM + (size_t)NT * NQP;
  float* spcA  = vArr + NQP;
  float* uArr  = spcA + NQP;
  int*   pathA = (int*)(uArr + NT);
  int*   remA  = pathA + NQP;
  int*   r4cA  = remA + NQP;
  int*   srA   = r4cA + NQP;
  int*   c4rA  = srA + NT;
  int*   rankA = c4rA + NT;
  float* tboxS = (float*)(rankA + NT);
  float* tangS = tboxS + NT * 4;
  float* redV  = tangS + NT;
  int*   redI  = (int*)(redV + 16);
  float* sMin  = (float*)(redI + 16);
  int*   sI    = (int*)(sMin + 1);
  int*   sSink = sI + 1;

  const int tid  = threadIdx.x;
  const int lane = tid & 31;
  const int wid  = tid >> 5;
  const int b    = blockIdx.x;

  // ---- stage target data into LDS (async-to-LDS path on CDNA5) ----
  const float* tb_g = g_tgt_boxes + (size_t)b * NT * 4;
  const float* ta_g = g_tgt_angle + (size_t)b * NT;
#if defined(USE_ASYNC_LDS)
  // builtin params are int-pointee ('__device__ int *'); pass int* args
  if (tid < NT * 4)
    __builtin_amdgcn_global_load_async_to_lds_b32(
        (int*)(tb_g + tid), (int*)(tboxS + tid), 0, 0);
  if (tid < NT)
    __builtin_amdgcn_global_load_async_to_lds_b32(
        (int*)(ta_g + tid), (int*)(tangS + tid), 0, 0);
  __builtin_amdgcn_s_wait_asynccnt(0);
#else
  if (tid < NT * 4) tboxS[tid] = tb_g[tid];
  if (tid < NT)     tangS[tid] = ta_g[tid];
#endif
  // init LAP state
  vArr[tid] = 0.0f;
  r4cA[tid] = -1;
  if (tid < NT) { uArr[tid] = 0.0f; c4rA[tid] = -1; }
  __syncthreads();

  // ---- build transposed cost matrix cost[t][q], resident in LDS ----
  const int q = tid;
  if (q < NQ) {
    const float* obp = g_out_boxes + ((size_t)b * NQ + q) * 4;
    float o0 = obp[0], o1 = obp[1], o2 = obp[2], o3 = obp[3];
    float o4 = ANG_MIN + g_out_angle[(size_t)b * NQ + q] * ANG_SPAN;
    float lg = g_out_logits[(size_t)b * NQ + q];
    float p  = 1.0f / (1.0f + expf(-lg));
    float lcost = -W_LOGITS * p;
    for (int t = 0; t < NT; ++t) {
      float s = fabsf(o0 - tboxS[t * 4 + 0]) + fabsf(o1 - tboxS[t * 4 + 1]) +
                fabsf(o2 - tboxS[t * 4 + 2]) + fabsf(o3 - tboxS[t * 4 + 3]) +
                fabsf(o4 - tangS[t]);
      costM[t * NQP + q] = W_BOXES * s + lcost;
    }
  } else {
    for (int t = 0; t < NT; ++t) costM[t * NQP + q] = PADCOST;  // pad cols
  }
  __syncthreads();

  // ---- shortest augmenting path LAP (rows = 64 targets, cols = 500 queries)
  for (int cur_row = 0; cur_row < NT; ++cur_row) {
    remA[tid]  = 1;
    spcA[tid]  = BIGS;
    pathA[tid] = -1;
    if (tid < NT) srA[tid] = 0;
    if (tid == 0) { *sI = cur_row; *sMin = 0.0f; *sSink = -1; }
    __syncthreads();

    while (true) {
      const int   i      = *sI;
      const float minval = *sMin;
      if (tid == 0) srA[i] = 1;
      float myv = BIGS;
      int   myj = tid;
      if (remA[tid]) {
        float r = minval + costM[i * NQP + tid] - uArr[i] - vArr[tid];
        float s = spcA[tid];
        if (r < s) { s = r; spcA[tid] = r; pathA[tid] = i; }
        myv = s;
      }
      // wave32 argmin (value, index) reduction
      red_pair(myv, myj, 16);
      red_pair(myv, myj, 8);
      red_pair(myv, myj, 4);
      red_pair(myv, myj, 2);
      red_pair(myv, myj, 1);
      if (lane == 0) { redV[wid] = myv; redI[wid] = myj; }
      __syncthreads();
      if (wid == 0) {
        float rv = (lane < 16) ? redV[lane] : BIGS;
        int   rj = (lane < 16) ? redI[lane] : 0x7fffffff;
        red_pair(rv, rj, 8);
        red_pair(rv, rj, 4);
        red_pair(rv, rj, 2);
        red_pair(rv, rj, 1);
        if (lane == 0) {
          *sMin = rv;
          remA[rj] = 0;
          int rc = r4cA[rj];
          if (rc < 0) *sSink = rj; else *sI = rc;
        }
      }
      __syncthreads();
      if (*sSink >= 0) break;
    }

    // dual updates (read pre-augment col4row, per scipy rectangular_lsap)
    const float minval = *sMin;
    const int   sink   = *sSink;
    if (tid < NT && srA[tid]) {
      if (tid == cur_row) {
        uArr[tid] += minval;
      } else {
        int scol = c4rA[tid]; if (scol < 0) scol = 0;
        uArr[tid] += minval - spcA[scol];
      }
    }
    if (!remA[tid]) vArr[tid] -= (minval - spcA[tid]);
    __syncthreads();

    // augment along predecessor path (cheap, serial)
    if (tid == 0) {
      int j = sink;
      while (true) {
        int i2 = pathA[j];
        r4cA[j] = i2;
        int jn = c4rA[i2];
        c4rA[i2] = j;
        if (i2 == cur_row) break;
        j = jn;
      }
    }
    __syncthreads();
  }

  // ---- argsort(col4row) via WMMA rank reduction: rank[m] = sum_k (c[k] < c[m])
  // A = 16x64 u8 comparison matrix, B = all-ones bytes -> D[m][*] = rank[m].
  if (tid < 32) {  // wave 0, EXEC all ones
    const int half = lane >> 4;
    const int mloc = lane & 15;
    v8i bOnes = {0x01010101, 0x01010101, 0x01010101, 0x01010101,
                 0x01010101, 0x01010101, 0x01010101, 0x01010101};
    v8i cZero = {0, 0, 0, 0, 0, 0, 0, 0};
#pragma unroll
    for (int tile = 0; tile < 4; ++tile) {
      const int m  = tile * 16 + mloc;
      const int cm = c4rA[m];
      v8i a;
#pragma unroll
      for (int r = 0; r < 8; ++r) {
        // 8-bit A-matrix 16x64 layout: VGPR r holds K = kBase..kBase+3
        const int kBase = (r >> 1) * 16 + half * 8 + (r & 1) * 4;
        unsigned pack = 0u;
#pragma unroll
        for (int bb = 0; bb < 4; ++bb)
          pack |= (unsigned)((c4rA[kBase + bb] < cm) ? 1 : 0) << (8 * bb);
        a[r] = (int)pack;
      }
      v8i d = __builtin_amdgcn_wmma_i32_16x16x64_iu8(false, a, false, bOnes,
                                                     cZero, false, false);
      if (mloc == 0) {  // lanes 0 (M=0..7) and 16 (M=8..15) of the tile
#pragma unroll
        for (int r = 0; r < 8; ++r) rankA[tile * 16 + half * 8 + r] = d[r];
      }
    }
  }
  __syncthreads();

  // scatter sorted output: out[b][0][r] = query, out[b][1][r] = target
  if (tid < NT) {
    const int rk = rankA[tid];
    int* outB = g_out + (size_t)b * 2 * NT;
    outB[rk]      = c4rA[tid];
    outB[NT + rk] = tid;
  }
}

extern "C" void kernel_launch(void* const* d_in, const int* in_sizes, int n_in,
                              void* d_out, int out_size, void* d_ws, size_t ws_size,
                              hipStream_t stream) {
  (void)in_sizes; (void)n_in; (void)out_size; (void)d_ws; (void)ws_size;
  const float* ob = (const float*)d_in[0];
  const float* oa = (const float*)d_in[1];
  const float* ol = (const float*)d_in[2];
  const float* tb = (const float*)d_in[3];
  const float* ta = (const float*)d_in[4];
  int* out = (int*)d_out;
  // ~144 KB dynamic LDS per block (CDNA5 WGP has 320 KB)
  (void)hipFuncSetAttribute(
      reinterpret_cast<const void*>(hungarian_matcher_kernel),
      hipFuncAttributeMaxDynamicSharedMemorySize, (int)SMEM_BYTES);
  hungarian_matcher_kernel<<<dim3(BS), dim3(BSZ), SMEM_BYTES, stream>>>(
      ob, oa, ol, tb, ta, out);
}
